// TemporalSA_83124797047414
// MI455X (gfx1250) — matrix operand order repaired and checked by reading, not compiled, not run on hardware
//
#include <hip/hip_runtime.h>

#define T_LEN 2048
#define CIN   512
#define B_N   4
#define EPS_F 1e-5f

typedef __attribute__((ext_vector_type(16))) __bf16       v16bf;
typedef __attribute__((ext_vector_type(8)))  float        v8f;
typedef __attribute__((ext_vector_type(4)))  unsigned int u32x4;

union V16 {
    v16bf v;
    u32x4 u4[2];
    unsigned short s[16];
};

__device__ __forceinline__ unsigned short f2bf(float f) {
    union { float f; unsigned u; } c; c.f = f;
    unsigned r = c.u + 0x7FFFu + ((c.u >> 16) & 1u);   // round-to-nearest-even
    return (unsigned short)(r >> 16);
}

// ---------------------------------------------------------------- prep kernels
__global__ void k_convert_x(const float* __restrict__ x, unsigned short* __restrict__ xT) {
    int idx = blockIdx.x * blockDim.x + threadIdx.x;            // over B*C*T
    if (idx >= B_N * CIN * T_LEN) return;
    int t = idx % T_LEN;
    int c = (idx / T_LEN) % CIN;
    int b = idx / (T_LEN * CIN);
    xT[((size_t)(b * T_LEN + t)) * CIN + c] = f2bf(x[idx]);
}

__global__ void k_convert_w(const float* __restrict__ Wq, const float* __restrict__ Wk,
                            const float* __restrict__ Wv, unsigned short* __restrict__ Wall) {
    int idx = blockIdx.x * blockDim.x + threadIdx.x;            // over 768*512
    if (idx >= 768 * 512) return;
    int c = idx & 511;
    int row = idx >> 9;
    float w;
    if (row < 128)      w = Wq[row * 512 + c];
    else if (row < 256) w = Wk[(row - 128) * 512 + c];
    else                w = Wv[(row - 256) * 512 + c];
    Wall[idx] = f2bf(w);
}

__global__ void k_scales(const float* bq, const float* gq, const float* betaq,
                         const float* bk, const float* gk, const float* betak,
                         const float* bv, const float* gv, const float* betav,
                         float* __restrict__ sc, float* __restrict__ bsv) {
    int o = blockIdx.x * blockDim.x + threadIdx.x;
    if (o >= 768) return;
    float g, bb, be;
    if (o < 128)      { g = gq[o];       bb = bq[o];       be = betaq[o];       }
    else if (o < 256) { g = gk[o - 128]; bb = bk[o - 128]; be = betak[o - 128]; }
    else              { g = gv[o - 256]; bb = bv[o - 256]; be = betav[o - 256]; }
    float inv = g * rsqrtf(1.0f + EPS_F);
    sc[o]  = inv;
    bsv[o] = bb * inv + be;
}

// ---------------------------------------------------------- projection (WMMA)
// y[b, o, t] = relu( (W[o,:] . x[b,:,t]) * sc[o] + bsv[o] )   o in [0,768)
// One wave computes a 16(o) x 64(t) register-blocked tile: the A (weight) tile
// is loaded once per K-step and reused by 4 independent WMMA accumulator
// chains (hides WMMA->WMMA RAW latency, cuts A traffic 4x).
__global__ __launch_bounds__(32)
void k_proj(const unsigned short* __restrict__ xT, const unsigned short* __restrict__ Wall,
            const float* __restrict__ sc, const float* __restrict__ bsv,
            unsigned short* __restrict__ qT, unsigned short* __restrict__ kT,
            unsigned short* __restrict__ vB) {
    const int lane = threadIdx.x & 31;
    const int lo = lane & 15, g = lane >> 4;
    const int t0 = blockIdx.x * 64;
    const int o0 = blockIdx.y * 16;
    const int b  = blockIdx.z;

    const unsigned short* wrow = Wall + (size_t)(o0 + lo) * 512;
    const unsigned short* xrow0 = xT + ((size_t)(b * T_LEN + t0 +  0 + lo)) * 512;
    const unsigned short* xrow1 = xT + ((size_t)(b * T_LEN + t0 + 16 + lo)) * 512;
    const unsigned short* xrow2 = xT + ((size_t)(b * T_LEN + t0 + 32 + lo)) * 512;
    const unsigned short* xrow3 = xT + ((size_t)(b * T_LEN + t0 + 48 + lo)) * 512;

    v8f acc0 = {}, acc1 = {}, acc2 = {}, acc3 = {};
    for (int kk = 0; kk < 16; ++kk) {
        const int cb = kk * 32;
        V16 a, b0, b1, b2, b3;
        // A (16x32 bf16): lane row = o0+lo; K = e + 16*g contiguous
        a.u4[0] = *(const u32x4*)(wrow + cb + 16 * g);
        a.u4[1] = *(const u32x4*)(wrow + cb + 16 * g + 8);
        // B (32x16 bf16): lane col = t; K = e + 16*g contiguous
        b0.u4[0] = *(const u32x4*)(xrow0 + cb + 16 * g);
        b0.u4[1] = *(const u32x4*)(xrow0 + cb + 16 * g + 8);
        b1.u4[0] = *(const u32x4*)(xrow1 + cb + 16 * g);
        b1.u4[1] = *(const u32x4*)(xrow1 + cb + 16 * g + 8);
        b2.u4[0] = *(const u32x4*)(xrow2 + cb + 16 * g);
        b2.u4[1] = *(const u32x4*)(xrow2 + cb + 16 * g + 8);
        b3.u4[0] = *(const u32x4*)(xrow3 + cb + 16 * g);
        b3.u4[1] = *(const u32x4*)(xrow3 + cb + 16 * g + 8);
        acc0 = __builtin_amdgcn_wmma_f32_16x16x32_bf16(false, a.v, false, b0.v,
                                                       (short)0, acc0, false, false);
        acc1 = __builtin_amdgcn_wmma_f32_16x16x32_bf16(false, a.v, false, b1.v,
                                                       (short)0, acc1, false, false);
        acc2 = __builtin_amdgcn_wmma_f32_16x16x32_bf16(false, a.v, false, b2.v,
                                                       (short)0, acc2, false, false);
        acc3 = __builtin_amdgcn_wmma_f32_16x16x32_bf16(false, a.v, false, b3.v,
                                                       (short)0, acc3, false, false);
    }

    // per-channel BN scale/bias + ReLU (same 16 o-rows for all 4 sub-tiles)
    float scv[8], bsvv[8];
    for (int r = 0; r < 8; ++r) {
        int o = o0 + r + 8 * g;                     // D row M = r + 8*g
        scv[r]  = sc[o];
        bsvv[r] = bsv[o];
    }
    v8f accs[4] = {acc0, acc1, acc2, acc3};
    for (int j = 0; j < 4; ++j) {
        float y[8];
        for (int r = 0; r < 8; ++r) {
            float yy = accs[j][r] * scv[r] + bsvv[r];
            y[r] = yy > 0.0f ? yy : 0.0f;
        }
        const int t = t0 + 16 * j + lo;             // D col N = lane&15
        if (o0 < 256) {
            unsigned short* base = (o0 < 128)
                ? (qT + ((size_t)(b * T_LEN + t)) * 128 + o0 + 8 * g)
                : (kT + ((size_t)(b * T_LEN + t)) * 128 + (o0 - 128) + 8 * g);
            union { u32x4 u; unsigned short s[8]; } pk;
            for (int r = 0; r < 8; ++r) pk.s[r] = f2bf(y[r]);
            *(u32x4*)base = pk.u;                   // 8 contiguous bf16
        } else {
            for (int r = 0; r < 8; ++r)
                vB[((size_t)(b * 512 + (o0 - 256) + r + 8 * g)) * T_LEN + t] = f2bf(y[r]);
        }
    }
}

// ---------------------------------------------------- sliding-window attention
// One wave per (b, h, 16-wide t tile). j band = [t0-32, t0+32), window d=j-t+31.
__global__ __launch_bounds__(32)
void k_attn(const unsigned short* __restrict__ qT, const unsigned short* __restrict__ kT,
            const unsigned short* __restrict__ vB, const float* __restrict__ x,
            float* __restrict__ out) {
    __shared__ __align__(16) float          Sl[16][52];   // [t][jrel 0..47]
    __shared__ __align__(16) unsigned short Pl[16][64];   // bf16 band matrix P'[t][jrel]

    const int lane = threadIdx.x & 31;
    const int lo = lane & 15, g = lane >> 4;
    const int t0 = blockIdx.x * 16;
    const int h  = blockIdx.y;
    const int b  = blockIdx.z;
    const u32x4 zz = {0u, 0u, 0u, 0u};

    // B operand (Q tile, 32c x 16t): lane col t = t0+lo, K = c contiguous
    V16 bq;
    const unsigned short* qrow = qT + ((size_t)(b * T_LEN + t0 + lo)) * 128 + h * 32;
    bq.u4[0] = *(const u32x4*)(qrow + 16 * g);
    bq.u4[1] = *(const u32x4*)(qrow + 16 * g + 8);

    const float rsc = 0.08838834764831845f;               // 1/sqrt(128)

    // ---- S^T = K^T * Q over three 16-wide j tiles -------------------------
    for (int jt = 0; jt < 3; ++jt) {
        const int jrow = t0 - 32 + jt * 16 + lo;          // A row = j
        V16 a;
        if (jrow >= 0) {
            const unsigned short* krow = kT + ((size_t)(b * T_LEN + jrow)) * 128 + h * 32;
            a.u4[0] = *(const u32x4*)(krow + 16 * g);
            a.u4[1] = *(const u32x4*)(krow + 16 * g + 8);
        } else {
            a.u4[0] = zz; a.u4[1] = zz;                    // causal zero padding
        }
        v8f s = {};
        s = __builtin_amdgcn_wmma_f32_16x16x32_bf16(false, a.v, false, bq.v,
                                                    (short)0, s, false, false);
        for (int r = 0; r < 8; ++r)
            Sl[lo][jt * 16 + r + 8 * g] = s[r] * rsc;      // row t=N, col jrel=M
    }
    __syncthreads();

    // ---- softmax over window d=0..31 (jrel = t_loc + 1 + d) ---------------
    float vals[16];
    float mx = -1e30f;
    for (int i = 0; i < 16; ++i) {
        vals[i] = Sl[lo][lo + 1 + 16 * g + i];
        mx = fmaxf(mx, vals[i]);
    }
    mx = fmaxf(mx, __shfl_xor(mx, 16, 32));
    float sum = 0.0f;
    for (int i = 0; i < 16; ++i) { vals[i] = __expf(vals[i] - mx); sum += vals[i]; }
    sum += __shfl_xor(sum, 16, 32);
    const float pinv = 1.0f / sum;

    // zero P' band matrix, then scatter probabilities into the band
    {
        u32x4* pz = (u32x4*)&Pl[lo][32 * g];
        pz[0] = zz; pz[1] = zz; pz[2] = zz; pz[3] = zz;
    }
    __syncthreads();
    for (int i = 0; i < 16; ++i)
        Pl[lo][lo + 1 + 16 * g + i] = f2bf(vals[i] * pinv);
    __syncthreads();

    // ---- att = V * P'  (dv=128 -> 8 c-tiles, K=64 -> 2 wmma chunks) -------
    for (int ct = 0; ct < 8; ++ct) {
        const int c = h * 128 + ct * 16 + lo;              // A row
        const unsigned short* vrow = vB + ((size_t)(b * 512 + c)) * T_LEN;
        v8f acc = {};
        for (int kc = 0; kc < 2; ++kc) {
            V16 a, p;
            const int j0 = t0 - 32 + 32 * kc + 8 * g;      // e0..7 run (8-aligned)
            const int j1 = j0 + 16;                        // e8..15 run
            a.u4[0] = (j0 >= 0 && j0 < T_LEN) ? *(const u32x4*)(vrow + j0) : zz;
            a.u4[1] = (j1 >= 0 && j1 < T_LEN) ? *(const u32x4*)(vrow + j1) : zz;
            p.u4[0] = *(const u32x4*)&Pl[lo][32 * kc + 8 * g];
            p.u4[1] = *(const u32x4*)&Pl[lo][32 * kc + 8 * g + 16];
            acc = __builtin_amdgcn_wmma_f32_16x16x32_bf16(false, a.v, false, p.v,
                                                          (short)0, acc, false, false);
        }
        const int t = t0 + lo;
        for (int r = 0; r < 8; ++r) {
            size_t oi = ((size_t)(b * 512 + h * 128 + ct * 16 + r + 8 * g)) * T_LEN + t;
            out[oi] = x[oi] + acc[r];                      // residual skip
        }
    }
}

// -------------------------------------------------------------------- launch
extern "C" void kernel_launch(void* const* d_in, const int* in_sizes, int n_in,
                              void* d_out, int out_size, void* d_ws, size_t ws_size,
                              hipStream_t stream) {
    const float* x     = (const float*)d_in[0];
    const float* Wq    = (const float*)d_in[1];
    const float* bq    = (const float*)d_in[2];
    const float* gq    = (const float*)d_in[3];
    const float* betaq = (const float*)d_in[4];
    const float* Wk    = (const float*)d_in[5];
    const float* bk    = (const float*)d_in[6];
    const float* gk    = (const float*)d_in[7];
    const float* betak = (const float*)d_in[8];
    const float* Wv    = (const float*)d_in[9];
    const float* bv    = (const float*)d_in[10];
    const float* gv    = (const float*)d_in[11];
    const float* betav = (const float*)d_in[12];

    char* ws = (char*)d_ws;
    unsigned short* xT   = (unsigned short*)(ws);                 //  8 MB  [B,T,512] bf16
    unsigned short* Wall = (unsigned short*)(ws +  8388608);      // 768 KB [768,512] bf16
    float*          sc   = (float*)         (ws +  9175040);      //   3 KB
    float*          bsv  = (float*)         (ws +  9178112);      //   3 KB
    unsigned short* qT   = (unsigned short*)(ws +  9181184);      //   2 MB [B,T,128] bf16
    unsigned short* kT   = (unsigned short*)(ws + 11278336);      //   2 MB [B,T,128] bf16
    unsigned short* vB   = (unsigned short*)(ws + 13375488);      //   8 MB [B,512,T] bf16

    k_convert_x<<<(B_N * CIN * T_LEN + 255) / 256, 256, 0, stream>>>(x, xT);
    k_convert_w<<<(768 * 512 + 255) / 256, 256, 0, stream>>>(Wq, Wk, Wv, Wall);
    k_scales<<<3, 256, 0, stream>>>(bq, gq, betaq, bk, gk, betak, bv, gv, betav, sc, bsv);

    // 768 output channels x 2048 t x 4 b, 16(o)x64(t) register-blocked tiles
    k_proj<<<dim3(T_LEN / 64, 768 / 16, B_N), 32, 0, stream>>>(xT, Wall, sc, bsv, qT, kT, vB);

    // per (t-tile, head, batch) banded attention
    k_attn<<<dim3(T_LEN / 16, 4, B_N), 32, 0, stream>>>(qT, kT, vB, x, (float*)d_out);
}
